// CachedMoEExperts_41540923687136
// MI455X (gfx1250) — matrix operand order repaired and checked by reading, not compiled, not run on hardware
//
#include <hip/hip_runtime.h>
#include <hip/hip_fp16.h>

typedef __attribute__((ext_vector_type(16))) _Float16 v16h;
typedef __attribute__((ext_vector_type(8)))  float    v8f;

#define NUM_EXPERTS 16
#define TOP_K 4
#define HIDDEN 2048
#define INTER 1408
#define TOKENS 4096

// ---------------- workspace layout (bytes) ----------------
constexpr size_t OFF_TOPK_IDX = 0;                                   // [T,4] int
constexpr size_t OFF_TOPK_W   = 64 * 1024;                           // [T,4] float
constexpr size_t OFF_COUNTS   = 128 * 1024;                          // [64] int
constexpr size_t OFF_LIST     = 132 * 1024;                          // [64][T] int
constexpr size_t OFF_LISTW    = OFF_LIST  + (size_t)64 * TOKENS * 4; // [64][T] float
constexpr size_t OFF_XH       = OFF_LISTW + (size_t)64 * TOKENS * 4; // [T,H] f16
constexpr size_t OFF_H        = OFF_XH + (size_t)TOKENS * HIDDEN * 2;// [T*4,I] f16

// ---------------- routing: softmax -> top-4 -> renormalize ----------------
__global__ void moe_router(const float* __restrict__ logits,
                           int* __restrict__ tidx, float* __restrict__ tw) {
    int t = blockIdx.x * blockDim.x + threadIdx.x;
    if (t >= TOKENS) return;
    float p[NUM_EXPERTS];
    float mx = -1e30f;
    #pragma unroll
    for (int e = 0; e < NUM_EXPERTS; ++e) { p[e] = logits[t * NUM_EXPERTS + e]; mx = fmaxf(mx, p[e]); }
    float s = 0.f;
    #pragma unroll
    for (int e = 0; e < NUM_EXPERTS; ++e) { p[e] = __expf(p[e] - mx); s += p[e]; }
    float inv = __builtin_amdgcn_rcpf(s);
    #pragma unroll
    for (int e = 0; e < NUM_EXPERTS; ++e) p[e] *= inv;
    int   idx[TOP_K]; float w[TOP_K]; float s4 = 0.f;
    #pragma unroll
    for (int k = 0; k < TOP_K; ++k) {
        int bi = 0; float bv = -1.f;
        #pragma unroll
        for (int e = 0; e < NUM_EXPERTS; ++e) if (p[e] > bv) { bv = p[e]; bi = e; }
        idx[k] = bi; w[k] = bv; p[bi] = -2.f; s4 += bv;
    }
    float r = __builtin_amdgcn_rcpf(s4);
    #pragma unroll
    for (int k = 0; k < TOP_K; ++k) { tidx[t * TOP_K + k] = idx[k]; tw[t * TOP_K + k] = w[k] * r; }
}

// ---------------- ordered per-(expert,slot) token list compaction ----------------
__global__ void moe_build_lists(const int* __restrict__ tidx, const float* __restrict__ tw,
                                int* __restrict__ lists, float* __restrict__ listw,
                                int* __restrict__ counts) {
    int bid = blockIdx.x;          // 0..63 : e*4+k
    int e = bid >> 2, k = bid & 3;
    int lane = threadIdx.x;        // wave32
    int cnt = 0;
    int*   L  = lists + (size_t)bid * TOKENS;
    float* LW = listw + (size_t)bid * TOKENS;
    for (int base = 0; base < TOKENS; base += 32) {
        int t = base + lane;
        bool pred = (tidx[t * TOP_K + k] == e);
        unsigned m = (unsigned)__ballot(pred);
        int ofs = __popc(m & ((1u << lane) - 1u));
        if (pred) { L[cnt + ofs] = t; LW[cnt + ofs] = tw[t * TOP_K + k]; }
        cnt += __popc(m);
    }
    int padded = (cnt + 127) & ~127;
    for (int i = cnt + lane; i < padded; i += 32) { L[i] = 0; LW[i] = 0.f; }
    if (lane == 0) counts[bid] = cnt;
}

// ---------------- x : f32 -> f16, 8 elements/thread ----------------
__global__ void moe_cvt_x(const float* __restrict__ x, _Float16* __restrict__ xh) {
    int i = blockIdx.x * blockDim.x + threadIdx.x;     // 8 floats each
    const float4* src = (const float4*)x + (size_t)i * 2;
    float4 f0 = src[0], f1 = src[1];
    union { _Float16 h[8]; uint4 u; } cv;
    cv.h[0] = (_Float16)f0.x; cv.h[1] = (_Float16)f0.y;
    cv.h[2] = (_Float16)f0.z; cv.h[3] = (_Float16)f0.w;
    cv.h[4] = (_Float16)f1.x; cv.h[5] = (_Float16)f1.y;
    cv.h[6] = (_Float16)f1.z; cv.h[7] = (_Float16)f1.w;
    ((uint4*)xh)[i] = cv.u;
}

// ---------------- WMMA GEMM core ----------------
// Tile: BM=128, BN=128, BK=32. 256 threads = 8 waves, each wave 32x64 (2x4 wmma tiles).
// Double-buffered LDS; A tile staged with GLOBAL_LOAD_ASYNC_TO_LDS_B128 (ASYNCcnt),
// B tile f32->f16 converted in regs with next-tile loads overlapping WMMA.
union AFrag { v16h h; uint4 u[2]; };

template <int KDIM, bool IS_GEMM1>
__global__ __launch_bounds__(256)
void moe_gemm(const _Float16* __restrict__ Ah,   // xh [T,H]  or  h [T*4,I]
              const float*    __restrict__ W,    // w1 [E,I,H] or w2 [E,H,I] (f32)
              _Float16*       __restrict__ Hout, // gemm1 output rows (f16, gate*silu)
              float*          __restrict__ Out,  // gemm2 output [T,H] f32
              const int*      __restrict__ lists,
              const float*    __restrict__ listw,
              const int*      __restrict__ counts,
              int kslot, int accum) {
    constexpr int BM = 128, BN = 128, BK = 32;
    constexpr int LDH = 48;                 // padded halves/row (96B, 16B-aligned chunks)

    int ek, e, kk;
    if (IS_GEMM1) { ek = blockIdx.z; e = ek >> 2; kk = ek & 3; }
    else          { e = blockIdx.z; kk = kslot;  ek = e * 4 + kk; }
    const int cnt = counts[ek];
    const int m_base = blockIdx.y * BM;
    if (m_base >= cnt) return;
    const int n_base = blockIdx.x * BN;

    const int*   list = lists + (size_t)ek * TOKENS;
    const float* lw   = listw + (size_t)ek * TOKENS;
    const float* Wexp = W + (size_t)e * ((size_t)INTER * HIDDEN);

    __shared__ __align__(16) unsigned short sA[2][BM * LDH];
    __shared__ __align__(16) unsigned short sB[2][BN * LDH];
    __shared__ int   sTok[BM];
    __shared__ float sG[BM];

    const int tid  = threadIdx.x;
    const int lane = tid & 31;
    const int wid  = tid >> 5;
    const int wm   = (wid & 3) * 32;   // wave M offset
    const int wn   = (wid >> 2) * 64;  // wave N offset
    const int l15  = lane & 15;
    const int hi   = lane >> 4;

    if (tid < BM) {
        int r = m_base + tid;
        sTok[tid] = list[r];           // lists are zero-padded to 128-multiples
        sG[tid]   = lw[r];
    }
    __syncthreads();

    // per-thread staging geometry
    const int brow  = tid >> 1;
    const int bhalf = (tid & 1) * 16;
    const float* Bbase = Wexp + (size_t)(n_base + brow) * KDIM + bhalf;

    auto stageA = [&](int b, int kb) {              // async gather: global f16 -> LDS
        #pragma unroll
        for (int i = 0; i < 2; ++i) {
            int ci  = tid + i * 256;                // 0..511 16B chunks
            int row = ci >> 2;
            int c8  = (ci & 3) * 8;
            int t   = sTok[row];
            size_t arow = IS_GEMM1 ? (size_t)t * HIDDEN : (size_t)(t * 4 + kk) * INTER;
            const _Float16* gp = Ah + arow + kb + c8;
            unsigned lo = (unsigned)(size_t)&sA[b][row * LDH + c8];
            asm volatile("global_load_async_to_lds_b128 %0, %1, off"
                         :: "v"(lo), "v"(gp) : "memory");
        }
    };
    auto loadB = [&](int kb, float4* f) {
        const float4* s = (const float4*)(Bbase + kb);
        f[0] = s[0]; f[1] = s[1]; f[2] = s[2]; f[3] = s[3];
        if (kb + BK < KDIM) __builtin_prefetch(Bbase + kb + BK, 0, 0);
    };
    auto commitB = [&](int b, const float4* f) {    // f32 -> f16, store to LDS
        union { _Float16 h[16]; uint4 u[2]; } cv;
        cv.h[0]  = (_Float16)f[0].x; cv.h[1]  = (_Float16)f[0].y;
        cv.h[2]  = (_Float16)f[0].z; cv.h[3]  = (_Float16)f[0].w;
        cv.h[4]  = (_Float16)f[1].x; cv.h[5]  = (_Float16)f[1].y;
        cv.h[6]  = (_Float16)f[1].z; cv.h[7]  = (_Float16)f[1].w;
        cv.h[8]  = (_Float16)f[2].x; cv.h[9]  = (_Float16)f[2].y;
        cv.h[10] = (_Float16)f[2].z; cv.h[11] = (_Float16)f[2].w;
        cv.h[12] = (_Float16)f[3].x; cv.h[13] = (_Float16)f[3].y;
        cv.h[14] = (_Float16)f[3].z; cv.h[15] = (_Float16)f[3].w;
        *(uint4*)&sB[b][brow * LDH + bhalf]     = cv.u[0];
        *(uint4*)&sB[b][brow * LDH + bhalf + 8] = cv.u[1];
    };

    v8f acc[2][4] = {};

    // prologue: stage k=0 into buffer 0
    {
        float4 f[4];
        loadB(0, f);
        stageA(0, 0);
        commitB(0, f);
    }
    asm volatile("s_wait_asynccnt 0x0" ::: "memory");
    __syncthreads();

    int buf = 0;
    for (int kb = 0; kb < KDIM; kb += BK) {
        const bool hasNext = (kb + BK < KDIM);
        float4 f[4] = {};
        if (hasNext) {
            loadB(kb + BK, f);           // global loads issue here, overlap WMMA
            stageA(buf ^ 1, kb + BK);    // async -> LDS, overlaps WMMA
        }

        // ---- fragments (ISA 16-bit A 16x32 layout; B split K over lane halves) ----
        AFrag a[2], b[4];
        #pragma unroll
        for (int mi = 0; mi < 2; ++mi) {
            int m = wm + mi * 16 + l15;
            a[mi].u[0] = *(const uint4*)&sA[buf][m * LDH + hi * 8];
            a[mi].u[1] = *(const uint4*)&sA[buf][m * LDH + 16 + hi * 8];
        }
        #pragma unroll
        for (int ni = 0; ni < 4; ++ni) {
            int n = wn + ni * 16 + l15;
            b[ni].u[0] = *(const uint4*)&sB[buf][n * LDH + hi * 8];
            b[ni].u[1] = *(const uint4*)&sB[buf][n * LDH + 16 + hi * 8];
        }
        #pragma unroll
        for (int mi = 0; mi < 2; ++mi)
            #pragma unroll
            for (int ni = 0; ni < 4; ++ni)
                acc[mi][ni] = __builtin_amdgcn_wmma_f32_16x16x32_f16(
                    false, a[mi].h, false, b[ni].h, (short)0, acc[mi][ni], false, false);

        if (hasNext) commitB(buf ^ 1, f);

        asm volatile("s_wait_asynccnt 0x0" ::: "memory");
        __syncthreads();
        buf ^= 1;
    }

    // ---- epilogue ----
    #pragma unroll
    for (int mi = 0; mi < 2; ++mi) {
        #pragma unroll
        for (int j = 0; j < 8; ++j) {
            int rloc = wm + mi * 16 + hi * 8 + j;
            int grow = m_base + rloc;
            if (grow < cnt) {
                int t = sTok[rloc];
                if (IS_GEMM1) {
                    float g = sG[rloc];
                    #pragma unroll
                    for (int ni = 0; ni < 4; ++ni) {
                        int n = n_base + wn + ni * 16 + l15;
                        float v = acc[mi][ni][j];
                        // silu via raw hardware reciprocal: v_rcp_f32 + v_mul
                        float s = v * __builtin_amdgcn_rcpf(1.f + __expf(-v));
                        Hout[(size_t)(t * 4 + kk) * INTER + n] = (_Float16)(s * g);
                    }
                } else {
                    #pragma unroll
                    for (int ni = 0; ni < 4; ++ni) {
                        int n = n_base + wn + ni * 16 + l15;
                        float v = acc[mi][ni][j];
                        size_t o = (size_t)t * HIDDEN + n;
                        Out[o] = accum ? (Out[o] + v) : v;
                    }
                }
            }
        }
    }
}

// ---------------- launcher ----------------
extern "C" void kernel_launch(void* const* d_in, const int* in_sizes, int n_in,
                              void* d_out, int out_size, void* d_ws, size_t ws_size,
                              hipStream_t stream) {
    const float* x      = (const float*)d_in[0];
    const float* logits = (const float*)d_in[1];
    const float* w1     = (const float*)d_in[2];
    const float* w2     = (const float*)d_in[3];
    float* out = (float*)d_out;

    char* ws = (char*)d_ws;
    int*       topk_idx = (int*)(ws + OFF_TOPK_IDX);
    float*     topk_w   = (float*)(ws + OFF_TOPK_W);
    int*       counts   = (int*)(ws + OFF_COUNTS);
    int*       lists    = (int*)(ws + OFF_LIST);
    float*     listw    = (float*)(ws + OFF_LISTW);
    _Float16*  xh       = (_Float16*)(ws + OFF_XH);
    _Float16*  hbuf     = (_Float16*)(ws + OFF_H);

    moe_router<<<TOKENS / 256, 256, 0, stream>>>(logits, topk_idx, topk_w);
    moe_build_lists<<<64, 32, 0, stream>>>(topk_idx, topk_w, lists, listw, counts);
    moe_cvt_x<<<(TOKENS * HIDDEN / 8) / 256, 256, 0, stream>>>(x, xh);

    dim3 g1(INTER / 128, TOKENS / 128, NUM_EXPERTS * TOP_K);
    moe_gemm<HIDDEN, true><<<g1, 256, 0, stream>>>(xh, w1, hbuf, nullptr,
                                                   lists, listw, counts, -1, 0);
    dim3 g2(HIDDEN / 128, TOKENS / 128, NUM_EXPERTS);
    for (int k = 0; k < TOP_K; ++k)
        moe_gemm<INTER, false><<<g2, 256, 0, stream>>>(hbuf, w2, nullptr, out,
                                                       lists, listw, counts, k, k > 0);
}
